// GlobalGraph_64269890617598
// MI455X (gfx1250) — matrix operand order repaired and checked by reading, not compiled, not run on hardware
//
#include <hip/hip_runtime.h>

// ---------------------------------------------------------------------------
// Problem constants (match reference)
// ---------------------------------------------------------------------------
static constexpr int N  = 50000;
static constexpr int E  = 800000;
static constexpr int C  = 64;     // in channels (= K dim of GEMMs)
static constexpr int HD = 128;    // H*D = q/k/v output width
static constexpr int EH = E * 2;  // edge*head items
static constexpr int NH = N * 2;  // node*head items

typedef __attribute__((ext_vector_type(16))) __bf16 v16bf;
typedef __attribute__((ext_vector_type(8)))  float  v8f;

// ---------------------------------------------------------------------------
// Kernel 1: fused projections  Q = xWq+bq, K = xWk+bk, V = xWv+bv  [N,128]
//           and skip           out = xWs+bs                        [N,64]
// via v_wmma_f32_16x16x32_bf16 with bf16x3 split for ~fp32 accuracy.
// One wave per 16x16 output tile. N = 3125*16 exactly (no remainder).
// ---------------------------------------------------------------------------
__device__ __forceinline__ void load_a_tile(const float* __restrict__ x,
                                            int row, int k0, int lane,
                                            v16bf& hi, v16bf& lo) {
  // 16-bit A 16x32 layout: lanes 0-15 -> K = k0 + {0..7, 16..23}
  //                        lanes16-31 -> K = k0 + {8..15, 24..31}
  const int kb = (lane & 16) ? 8 : 0;
  const float* p = x + row * C + k0 + kb;
#pragma unroll
  for (int e = 0; e < 16; ++e) {
    const int K = (e < 8) ? e : (e + 8);
    float f = p[K];
    __bf16 h = (__bf16)f;
    hi[e] = h;
    lo[e] = (__bf16)(f - (float)h);
  }
}

__device__ __forceinline__ void load_b_tile(const float* __restrict__ W,
                                            int ld, int col0, int k0, int lane,
                                            v16bf& hi, v16bf& lo) {
  // 16-bit B 32x16 layout: lanes 0-15 -> K = k0 + {0..7, 16..23}
  //                        lanes16-31 -> K = k0 + {8..15, 24..31}
  const int kb = (lane & 16) ? 8 : 0;
  const int n  = col0 + (lane & 15);
  const float* p = W + (size_t)(k0 + kb) * ld + n;
#pragma unroll
  for (int e = 0; e < 16; ++e) {
    const int K = (e < 8) ? e : (e + 8);
    float f = p[(size_t)K * ld];
    __bf16 h = (__bf16)f;
    hi[e] = h;
    lo[e] = (__bf16)(f - (float)h);
  }
}

__global__ void proj_wmma_kernel(const float* __restrict__ x,
                                 const float* __restrict__ Wq, const float* __restrict__ bq,
                                 const float* __restrict__ Wk, const float* __restrict__ bk,
                                 const float* __restrict__ Wv, const float* __restrict__ bv,
                                 const float* __restrict__ Ws, const float* __restrict__ bs,
                                 float* __restrict__ Qb, float* __restrict__ Kb,
                                 float* __restrict__ Vb, float* __restrict__ outp) {
  const int ROW_TILES = N / 16;        // 3125
  const int COL_TILES = 8 + 8 + 8 + 4; // Wq,Wk,Wv (128 cols) + Ws (64 cols)

  const int wave = (int)((blockIdx.x * blockDim.x + threadIdx.x) >> 5);
  const int lane = (int)(threadIdx.x & 31);
  if (wave >= ROW_TILES * COL_TILES) return;   // wave-uniform exit

  const int rt = wave / COL_TILES;
  const int ct = wave % COL_TILES;

  const float* W;  const float* bias;  float* dstp;  int ld;  int col0;
  if (ct < 8)       { W = Wq; bias = bq; dstp = Qb;   ld = HD; col0 = ct * 16;        }
  else if (ct < 16) { W = Wk; bias = bk; dstp = Kb;   ld = HD; col0 = (ct - 8)  * 16; }
  else if (ct < 24) { W = Wv; bias = bv; dstp = Vb;   ld = HD; col0 = (ct - 16) * 16; }
  else              { W = Ws; bias = bs; dstp = outp; ld = C;  col0 = (ct - 24) * 16; }

  // Accumulator initialized with bias: D(m,n) has n = lane&15 for all 8 elems.
  const float b = bias[col0 + (lane & 15)];
  v8f c;
#pragma unroll
  for (int i = 0; i < 8; ++i) c[i] = b;

  const int row = rt * 16 + (lane & 15);  // A-matrix row for this lane

#pragma unroll
  for (int ks = 0; ks < 2; ++ks) {        // K = 64 = 2 x 32
    const int k0 = ks * 32;
    v16bf a_hi, a_lo, b_hi, b_lo;
    load_a_tile(x, row, k0, lane, a_hi, a_lo);
    load_b_tile(W, ld, col0, k0, lane, b_hi, b_lo);
    // bf16x3: a*b ~= ah*bh + ah*bl + al*bh  (f32 accumulate)
    c = __builtin_amdgcn_wmma_f32_16x16x32_bf16(false, a_hi, false, b_hi,
                                                (short)0, c, false, false);
    c = __builtin_amdgcn_wmma_f32_16x16x32_bf16(false, a_hi, false, b_lo,
                                                (short)0, c, false, false);
    c = __builtin_amdgcn_wmma_f32_16x16x32_bf16(false, a_lo, false, b_hi,
                                                (short)0, c, false, false);
  }

  // Store D: VGPR i -> row rt*16 + i (+8 for upper half-wave), col = col0+(lane&15)
  const int n     = col0 + (lane & 15);
  const int mBase = rt * 16 + ((lane & 16) ? 8 : 0);
#pragma unroll
  for (int i = 0; i < 8; ++i)
    dstp[(size_t)(mBase + i) * ld + n] = c[i];
}

// ---------------------------------------------------------------------------
// Kernel 0: zero per-(node,head) max-key and denom
// ---------------------------------------------------------------------------
__global__ void init_kernel(unsigned* __restrict__ maxkey, float* __restrict__ denom) {
  int t = (int)(blockIdx.x * blockDim.x + threadIdx.x);
  if (t < NH) { maxkey[t] = 0u; denom[t] = 0.0f; }
}

// Order-preserving f32 <-> u32 encoding for atomic max over signed floats.
__device__ __forceinline__ unsigned enc_f32(float f) {
  unsigned u = __float_as_uint(f);
  return (u & 0x80000000u) ? ~u : (u | 0x80000000u);
}
__device__ __forceinline__ float dec_f32(unsigned k) {
  unsigned u = (k & 0x80000000u) ? (k ^ 0x80000000u) : ~k;
  return __uint_as_float(u);
}

// ---------------------------------------------------------------------------
// Kernel 2: per-(edge,head) logit = (q_dst . k_src)/8, store + segment-max
// ---------------------------------------------------------------------------
__global__ void logits_kernel(const float* __restrict__ Qb, const float* __restrict__ Kb,
                              const int* __restrict__ src, const int* __restrict__ dst,
                              float* __restrict__ logits, unsigned* __restrict__ maxkey) {
  int t = (int)(blockIdx.x * blockDim.x + threadIdx.x);
  if (t >= EH) return;
  const int e = t >> 1, h = t & 1;
  const int s = src[e], d = dst[e];
  const float4* qp = (const float4*)(Qb + (size_t)d * HD + h * 64);
  const float4* kp = (const float4*)(Kb + (size_t)s * HD + h * 64);
  float acc = 0.0f;
#pragma unroll
  for (int i = 0; i < 16; ++i) {
    float4 a = qp[i], b = kp[i];
    acc += a.x * b.x + a.y * b.y + a.z * b.z + a.w * b.w;
  }
  const float lg = acc * 0.125f;  // 1/sqrt(64)
  logits[t] = lg;
  atomicMax(&maxkey[d * 2 + h], enc_f32(lg));
}

// ---------------------------------------------------------------------------
// Kernel 3: ex = exp(logit - m[dst]); denom[dst] += ex  (overwrite logits buf)
// ---------------------------------------------------------------------------
__global__ void expsum_kernel(const int* __restrict__ dst,
                              float* __restrict__ logits,
                              const unsigned* __restrict__ maxkey,
                              float* __restrict__ denom) {
  int t = (int)(blockIdx.x * blockDim.x + threadIdx.x);
  if (t >= EH) return;
  const int e = t >> 1, h = t & 1;
  const int d = dst[e];
  const float m  = dec_f32(maxkey[d * 2 + h]);  // node has >=1 edge here
  const float ex = __expf(logits[t] - m);
  logits[t] = ex;
  atomicAdd(&denom[d * 2 + h], ex);
}

// ---------------------------------------------------------------------------
// Kernel 4: out[dst] += 0.5 * alpha * v[src]  (head-mean folded into 0.5)
// One wave per (edge, head); lane covers dims {lane, lane+32}.
// ---------------------------------------------------------------------------
__global__ void scatter_kernel(const float* __restrict__ Vb,
                               const int* __restrict__ src, const int* __restrict__ dst,
                               const float* __restrict__ ex, const float* __restrict__ denom,
                               float* __restrict__ outp) {
  const int w    = (int)((blockIdx.x * blockDim.x + threadIdx.x) >> 5);
  const int lane = (int)(threadIdx.x & 31);
  if (w >= EH) return;
  const int e = w >> 1, h = w & 1;
  const int s = src[e], d = dst[e];
  const float alpha = ex[w] / (denom[d * 2 + h] + 1e-16f);
  const float scale = 0.5f * alpha;
  const float* vp = Vb + (size_t)s * HD + h * 64;
  float* op = outp + (size_t)d * C;
  atomicAdd(&op[lane],      scale * vp[lane]);
  atomicAdd(&op[lane + 32], scale * vp[lane + 32]);
}

// ---------------------------------------------------------------------------
// Host launch
// ---------------------------------------------------------------------------
extern "C" void kernel_launch(void* const* d_in, const int* in_sizes, int n_in,
                              void* d_out, int out_size, void* d_ws, size_t ws_size,
                              hipStream_t stream) {
  (void)in_sizes; (void)n_in; (void)out_size; (void)ws_size;

  const float* x   = (const float*)d_in[0];
  const int*   ei  = (const int*)  d_in[1];   // [2, E]
  const float* Wq  = (const float*)d_in[2];
  const float* bq  = (const float*)d_in[3];
  const float* Wk  = (const float*)d_in[4];
  const float* bk  = (const float*)d_in[5];
  const float* Wv  = (const float*)d_in[6];
  const float* bv  = (const float*)d_in[7];
  const float* Ws  = (const float*)d_in[8];
  const float* bs  = (const float*)d_in[9];
  float* outp = (float*)d_out;

  const int* src = ei;       // edge_indices[0]
  const int* dst = ei + E;   // edge_indices[1]

  // Workspace layout (bytes, sections packed back to back)
  char* ws = (char*)d_ws;
  float*    Qb     = (float*)   (ws);                               // N*128 f32
  float*    Kb     = (float*)   (ws + (size_t)N * HD * 4);          // N*128 f32
  float*    Vb     = (float*)   (ws + (size_t)2 * N * HD * 4);      // N*128 f32
  float*    lg     = (float*)   (ws + (size_t)3 * N * HD * 4);      // EH f32
  unsigned* maxkey = (unsigned*)(ws + (size_t)3 * N * HD * 4 + (size_t)EH * 4);
  float*    denom  = (float*)   (ws + (size_t)3 * N * HD * 4 + (size_t)EH * 4
                                    + (size_t)NH * 4);

  // 0) zero segment-max keys and denominators
  init_kernel<<<(NH + 255) / 256, 256, 0, stream>>>(maxkey, denom);

  // 1) WMMA projections (also writes skip term into d_out)
  {
    const int waves  = (N / 16) * 28;              // 87500
    const int blocks = (waves + 7) / 8;            // 8 waves / 256-thread block
    proj_wmma_kernel<<<blocks, 256, 0, stream>>>(x, Wq, bq, Wk, bk, Wv, bv, Ws, bs,
                                                 Qb, Kb, Vb, outp);
  }

  // 2) edge logits + segment max
  logits_kernel<<<(EH + 255) / 256, 256, 0, stream>>>(Qb, Kb, src, dst, lg, maxkey);

  // 3) exp + segment sum
  expsum_kernel<<<(EH + 255) / 256, 256, 0, stream>>>(dst, lg, maxkey, denom);

  // 4) weighted scatter into out (on top of skip)
  {
    const long long threads = (long long)EH * 32;
    const int blocks = (int)((threads + 255) / 256);
    scatter_kernel<<<blocks, 256, 0, stream>>>(Vb, src, dst, lg, denom, outp);
  }
}
